// NormalizedEightPointNet_2783138808085
// MI455X (gfx1250) — matrix operand order repaired and checked by reading, not compiled, not run on hardware
//
#include <hip/hip_runtime.h>
#include <hip/hip_bf16.h>

// ---------------------------------------------------------------------------
// NormalizedEightPointNet forward for MI455X (gfx1250, wave32, WMMA).
// MLP (104 GFLOP) runs as bf16 WMMA GEMMs (v_wmma_f32_16x16x32_bf16) with
// 2x4-tile register blocking (8 acc tiles / wave); instance-norm uses
// 16-channel coalesced blocks. Everything else is cheap reductions / tiny
// Jacobi eigen solves.
// ---------------------------------------------------------------------------

typedef __attribute__((ext_vector_type(16))) __bf16 v16bf;
typedef __attribute__((ext_vector_type(8)))  __bf16 v8bf;
typedef __attribute__((ext_vector_type(8)))  float  v8f;

#define NB 16
#define NP 4096
#define MT (NB * NP)   // 65536 total points

// ---- bf16 helpers (RNE float->bf16) ---------------------------------------
__device__ __forceinline__ __bf16 f2bf(float f) {
    union { float f; unsigned u; } c; c.f = f;
    unsigned r = c.u + 0x7FFFu + ((c.u >> 16) & 1u);
    unsigned short h = (unsigned short)(r >> 16);
    __bf16 b; __builtin_memcpy(&b, &h, 2); return b;
}
__device__ __forceinline__ float bf2f(__bf16 b) {
    unsigned short h; __builtin_memcpy(&h, &b, 2);
    union { unsigned u; float f; } c; c.u = ((unsigned)h) << 16; return c.f;
}

// ---- block reductions (blockDim.x == 256) ---------------------------------
__device__ __forceinline__ float block_sum(float v, float* red) {
    int t = threadIdx.x; red[t] = v; __syncthreads();
    for (int s = 128; s > 0; s >>= 1) { if (t < s) red[t] += red[t + s]; __syncthreads(); }
    float r = red[0]; __syncthreads(); return r;
}
__device__ __forceinline__ float block_max(float v, float* red) {
    int t = threadIdx.x; red[t] = v; __syncthreads();
    for (int s = 128; s > 0; s >>= 1) { if (t < s) red[t] = fmaxf(red[t], red[t + s]); __syncthreads(); }
    float r = red[0]; __syncthreads(); return r;
}

// ---------------------------------------------------------------------------
// 0) fp32 -> bf16 weight conversion
// ---------------------------------------------------------------------------
__global__ void cvt_bf16_kernel(const float* __restrict__ src, __bf16* __restrict__ dst, int n) {
    int i = blockIdx.x * 256 + threadIdx.x;
    if (i < n) dst[i] = f2bf(src[i]);
}

// ---------------------------------------------------------------------------
// 1) plain normalization per batch: T1/T2 + normalized 2D points
// ---------------------------------------------------------------------------
__global__ void __launch_bounds__(256) prep_kernel(const float* __restrict__ pts,
                                                   float* __restrict__ p1n, float* __restrict__ p2n,
                                                   float* __restrict__ T1, float* __restrict__ T2) {
    __shared__ float red[256];
    int b = blockIdx.x, tid = threadIdx.x;
    const float* pb = pts + (size_t)b * NP * 4;
    float sx1 = 0.f, sy1 = 0.f, sx2 = 0.f, sy2 = 0.f;
    for (int n = tid; n < NP; n += 256) {
        const float* p = pb + n * 4;
        sx1 += p[0]; sy1 += p[1]; sx2 += p[2]; sy2 += p[3];
    }
    float c1x = block_sum(sx1, red) * (1.f / NP);
    float c1y = block_sum(sy1, red) * (1.f / NP);
    float c2x = block_sum(sx2, red) * (1.f / NP);
    float c2y = block_sum(sy2, red) * (1.f / NP);
    float d1 = 0.f, d2 = 0.f;
    for (int n = tid; n < NP; n += 256) {
        const float* p = pb + n * 4;
        float dx1 = p[0] - c1x, dy1 = p[1] - c1y;
        float dx2 = p[2] - c2x, dy2 = p[3] - c2y;
        d1 += sqrtf(dx1 * dx1 + dy1 * dy1);
        d2 += sqrtf(dx2 * dx2 + dy2 * dy2);
    }
    float md1 = block_sum(d1, red) * (1.f / NP);
    float md2 = block_sum(d2, red) * (1.f / NP);
    float s1 = 1.f / md1, s2 = 1.f / md2;
    if (tid == 0) {
        float* t1 = T1 + b * 9; float* t2 = T2 + b * 9;
        t1[0] = s1; t1[1] = 0.f; t1[2] = -c1x * s1;
        t1[3] = 0.f; t1[4] = s1; t1[5] = -c1y * s1;
        t1[6] = 0.f; t1[7] = 0.f; t1[8] = 1.f;
        t2[0] = s2; t2[1] = 0.f; t2[2] = -c2x * s2;
        t2[3] = 0.f; t2[4] = s2; t2[5] = -c2y * s2;
        t2[6] = 0.f; t2[7] = 0.f; t2[8] = 1.f;
    }
    for (int n = tid; n < NP; n += 256) {
        const float* p = pb + n * 4;
        size_t m = (size_t)b * NP + n;
        p1n[m * 2 + 0] = s1 * (p[0] - c1x);
        p1n[m * 2 + 1] = s1 * (p[1] - c1y);
        p2n[m * 2 + 0] = s2 * (p[2] - c2x);
        p2n[m * 2 + 1] = s2 * (p[3] - c2y);
    }
}

// ---------------------------------------------------------------------------
// 2) layer 0: 6 -> 64, VALU (K too small for WMMA)
// ---------------------------------------------------------------------------
__global__ void __launch_bounds__(256) layer0_kernel(const float* __restrict__ p1n,
                                                     const float* __restrict__ p2n,
                                                     const float* __restrict__ side,
                                                     const float* __restrict__ w0,
                                                     const float* __restrict__ b0,
                                                     __bf16* __restrict__ a0) {
    size_t m = (size_t)blockIdx.x * 256 + threadIdx.x;
    float in[6];
    in[0] = (p1n[m * 2 + 0] + 1.f) * 0.5f;
    in[1] = (p1n[m * 2 + 1] + 1.f) * 0.5f;
    in[2] = (p2n[m * 2 + 0] + 1.f) * 0.5f;
    in[3] = (p2n[m * 2 + 1] + 1.f) * 0.5f;
    in[4] = side[m * 2 + 0];
    in[5] = side[m * 2 + 1];
    for (int o = 0; o < 64; ++o) {
        float acc = b0[o];
        #pragma unroll
        for (int c = 0; c < 6; ++c) acc += w0[o * 6 + c] * in[c];
        a0[m * 64 + o] = f2bf(acc);
    }
}

// ---------------------------------------------------------------------------
// 3) bf16 WMMA GEMM: Y[m,o] = sum_k A[m,k] * W[o,k]   (bias dropped; cancels
//    in the instance norm that always follows).
//    One wave computes a 32(M) x 64(O) region = 2x4 tiles of 16x16:
//    8 WMMA per k-step for 12 b128 loads (vs 4 loads/WMMA unblocked).
//    Fragment layouts (A/B use this code's own contiguous K order, same for both):
//      A 16x32: lane L -> row m0+(L&15); halves = K {grp*16..+15}
//      B 32x16: lane L -> col o0+(L&15); halves = K {grp*16..+15}
//      D: lane L -> col o0+(L&15), VGPR v -> row m0 + grp*8 + v
// ---------------------------------------------------------------------------
__global__ void __launch_bounds__(256) gemm_bf16_kernel(const __bf16* __restrict__ A,
                                                        const __bf16* __restrict__ W,
                                                        __bf16* __restrict__ Y,
                                                        int K, int O) {
    int wave = threadIdx.x >> 5;
    int lane = threadIdx.x & 31;
    int oblocks = O >> 6;                       // 64-wide O blocks
    int wid = blockIdx.x * 8 + wave;
    int mb = wid / oblocks, ob = wid % oblocks;
    int m0 = mb << 5, o0 = ob << 6;
    int grp = lane >> 4, lm = lane & 15;

    const __bf16* arow = A + (size_t)(m0 + lm) * K + grp * 16;
    const __bf16* wrow = W + (size_t)(o0 + lm) * K + grp * 16;
    const size_t tstride = (size_t)16 * K;      // next 16-row tile

    v8f acc[8] = {};                            // acc[i*4+j]: A-tile i, B-tile j

    for (int k0 = 0; k0 < K; k0 += 32) {
        v16bf a[2], b[4];
        #pragma unroll
        for (int i = 0; i < 2; ++i) {
            v8bf lo = *(const v8bf*)(arow + i * tstride + k0);
            v8bf hi = *(const v8bf*)(arow + i * tstride + k0 + 8);
            #pragma unroll
            for (int e = 0; e < 8; ++e) { a[i][e] = lo[e]; a[i][e + 8] = hi[e]; }
        }
        #pragma unroll
        for (int j = 0; j < 4; ++j) {
            v8bf lo = *(const v8bf*)(wrow + j * tstride + k0);
            v8bf hi = *(const v8bf*)(wrow + j * tstride + k0 + 8);
            #pragma unroll
            for (int e = 0; e < 8; ++e) { b[j][e] = lo[e]; b[j][e + 8] = hi[e]; }
        }
        #pragma unroll
        for (int i = 0; i < 2; ++i)
            #pragma unroll
            for (int j = 0; j < 4; ++j)
                acc[i * 4 + j] = __builtin_amdgcn_wmma_f32_16x16x32_bf16(
                    false, a[i], false, b[j], (short)0, acc[i * 4 + j], false, false);
    }
    #pragma unroll
    for (int i = 0; i < 2; ++i) {
        __bf16* yb = Y + (size_t)(m0 + i * 16 + grp * 8) * O + lm;
        #pragma unroll
        for (int j = 0; j < 4; ++j)
            #pragma unroll
            for (int v = 0; v < 8; ++v)
                yb[(size_t)v * O + o0 + j * 16] = f2bf(acc[i * 4 + j][v]);
    }
}

// ---------------------------------------------------------------------------
// 4) instance norm + leaky ReLU, in place on bf16.
//    Block = (batch, 16-channel tile); thread t -> channel t&15, n-lane t>>4,
//    so 16 consecutive lanes touch 32 contiguous bytes (coalesced).
// ---------------------------------------------------------------------------
__global__ void __launch_bounds__(256) instnorm_kernel(__bf16* __restrict__ a,
                                                       const float* __restrict__ g,
                                                       const float* __restrict__ be,
                                                       int O) {
    __shared__ float red[256];
    __shared__ float smu[16], srs[16];
    int otiles = O >> 4;
    int b  = blockIdx.x / otiles;
    int o0 = (blockIdx.x % otiles) << 4;
    int t  = threadIdx.x;
    int oc = t & 15;          // channel within tile
    int nl = t >> 4;          // 16 n-lanes
    size_t base = (size_t)b * NP * O + o0 + oc;

    float s = 0.f, s2 = 0.f;
    for (int n = nl; n < NP; n += 16) {
        float v = bf2f(a[base + (size_t)n * O]);
        s += v; s2 += v * v;
    }
    red[t] = s; __syncthreads();
    for (int st = 8; st > 0; st >>= 1) { if (nl < st) red[t] += red[t + (st << 4)]; __syncthreads(); }
    if (nl == 0) smu[oc] = red[oc] * (1.f / NP);
    __syncthreads();
    red[t] = s2; __syncthreads();
    for (int st = 8; st > 0; st >>= 1) { if (nl < st) red[t] += red[t + (st << 4)]; __syncthreads(); }
    if (nl == 0) {
        float mu = smu[oc];
        float var = red[oc] * (1.f / NP) - mu * mu;
        srs[oc] = 1.f / sqrtf(var + 1e-5f);
    }
    __syncthreads();
    float mu = smu[oc], rs = srs[oc];
    float gg = g[o0 + oc], bb = be[o0 + oc];
    for (int n = nl; n < NP; n += 16) {
        float v = bf2f(a[base + (size_t)n * O]);
        float x = gg * (v - mu) * rs + bb;
        x = (x > 0.f) ? x : 0.01f * x;          // leaky_relu, slope 0.01
        a[base + (size_t)n * O] = f2bf(x);
    }
}

// ---------------------------------------------------------------------------
// 5) layer 5: 256 -> 1 logits; one wave per point, shuffle reduction
// ---------------------------------------------------------------------------
__global__ void __launch_bounds__(256) layer5_kernel(const __bf16* __restrict__ a4,
                                                     const float* __restrict__ w5,
                                                     const float* __restrict__ b5,
                                                     float* __restrict__ logits) {
    int wave = threadIdx.x >> 5, lane = threadIdx.x & 31;
    size_t m = (size_t)blockIdx.x * 8 + wave;
    const __bf16* row = a4 + m * 256 + lane * 8;
    float acc = 0.f;
    #pragma unroll
    for (int i = 0; i < 8; ++i) acc += bf2f(row[i]) * w5[lane * 8 + i];
    #pragma unroll
    for (int s = 16; s > 0; s >>= 1) acc += __shfl_xor(acc, s, 32);
    if (lane == 0) logits[m] = acc + b5[0];
}

// ---------------------------------------------------------------------------
// 6) softmax over N per batch -> weights output
// ---------------------------------------------------------------------------
__global__ void __launch_bounds__(256) softmax_kernel(const float* __restrict__ logits,
                                                      float* __restrict__ wout) {
    __shared__ float red[256];
    int b = blockIdx.x, tid = threadIdx.x;
    const float* l = logits + (size_t)b * NP;
    float mx = -3.4e38f;
    for (int n = tid; n < NP; n += 256) mx = fmaxf(mx, l[n]);
    mx = block_max(mx, red);
    float s = 0.f;
    for (int n = tid; n < NP; n += 256) s += expf(l[n] - mx);
    s = block_sum(s, red);
    float inv = 1.f / s;
    for (int n = tid; n < NP; n += 256) wout[(size_t)b * NP + n] = expf(l[n] - mx) * inv;
}

// ---------------------------------------------------------------------------
// 7) weighted normalization parameters per batch
// ---------------------------------------------------------------------------
__global__ void __launch_bounds__(256) wnorm_kernel(const float* __restrict__ p1n,
                                                    const float* __restrict__ p2n,
                                                    const float* __restrict__ w,
                                                    float* __restrict__ tw) {
    __shared__ float red[256];
    int b = blockIdx.x, tid = threadIdx.x;
    size_t base = (size_t)b * NP;
    float sw = 0.f, sx1 = 0.f, sy1 = 0.f, sx2 = 0.f, sy2 = 0.f;
    for (int n = tid; n < NP; n += 256) {
        float wn = w[base + n];
        sw  += wn;
        sx1 += wn * p1n[(base + n) * 2 + 0];
        sy1 += wn * p1n[(base + n) * 2 + 1];
        sx2 += wn * p2n[(base + n) * 2 + 0];
        sy2 += wn * p2n[(base + n) * 2 + 1];
    }
    float denom = block_sum(sw, red);
    float c1x = block_sum(sx1, red) / denom;
    float c1y = block_sum(sy1, red) / denom;
    float c2x = block_sum(sx2, red) / denom;
    float c2y = block_sum(sy2, red) / denom;
    float d1 = 0.f, d2 = 0.f;
    for (int n = tid; n < NP; n += 256) {
        float wn = w[base + n];
        float dx1 = p1n[(base + n) * 2 + 0] - c1x, dy1 = p1n[(base + n) * 2 + 1] - c1y;
        float dx2 = p2n[(base + n) * 2 + 0] - c2x, dy2 = p2n[(base + n) * 2 + 1] - c2y;
        d1 += wn * sqrtf(dx1 * dx1 + dy1 * dy1);
        d2 += wn * sqrtf(dx2 * dx2 + dy2 * dy2);
    }
    float md1 = block_sum(d1, red) / denom;
    float md2 = block_sum(d2, red) / denom;
    if (tid == 0) {
        float* t = tw + b * 8;
        t[0] = 1.4142f / md1; t[1] = c1x; t[2] = c1y;
        t[3] = 1.4142f / md2; t[4] = c2x; t[5] = c2y;
    }
}

// ---------------------------------------------------------------------------
// 8) XtX = sum_n w^2 * p p^T   (9x9 per batch, block-local reduce)
// ---------------------------------------------------------------------------
__global__ void __launch_bounds__(256) xtx_kernel(const float* __restrict__ p1n,
                                                  const float* __restrict__ p2n,
                                                  const float* __restrict__ w,
                                                  const float* __restrict__ tw,
                                                  float* __restrict__ xtx) {
    __shared__ float red[256];
    int b = blockIdx.x, tid = threadIdx.x;
    size_t base = (size_t)b * NP;
    const float* t = tw + b * 8;
    float s1 = t[0], c1x = t[1], c1y = t[2];
    float s2 = t[3], c2x = t[4], c2y = t[5];
    float acc[45];
    #pragma unroll
    for (int e = 0; e < 45; ++e) acc[e] = 0.f;
    for (int n = tid; n < NP; n += 256) {
        float wn = w[base + n]; float w2 = wn * wn;
        float ax = s1 * (p1n[(base + n) * 2 + 0] - c1x);
        float ay = s1 * (p1n[(base + n) * 2 + 1] - c1y);
        float bx = s2 * (p2n[(base + n) * 2 + 0] - c2x);
        float by = s2 * (p2n[(base + n) * 2 + 1] - c2y);
        float p[9] = { ax * bx, ax * by, ax,
                       ay * bx, ay * by, ay,
                       bx,      by,      1.f };
        int e = 0;
        #pragma unroll
        for (int i = 0; i < 9; ++i)
            #pragma unroll
            for (int j = i; j < 9; ++j) acc[e++] += w2 * p[i] * p[j];
    }
    int e = 0;
    for (int i = 0; i < 9; ++i)
        for (int j = i; j < 9; ++j) {
            float v = block_sum(acc[e], red);
            if (tid == 0) { xtx[b * 81 + i * 9 + j] = v; xtx[b * 81 + j * 9 + i] = v; }
            e++;
        }
}

// ---------------------------------------------------------------------------
// 9) tiny eigen solves + final transform, one thread per batch
// ---------------------------------------------------------------------------
__device__ void jacobi_sym(float* A, float* V, int n, int sweeps) {
    for (int i = 0; i < n * n; ++i) V[i] = 0.f;
    for (int i = 0; i < n; ++i) V[i * n + i] = 1.f;
    for (int s = 0; s < sweeps; ++s)
        for (int p = 0; p < n - 1; ++p)
            for (int q = p + 1; q < n; ++q) {
                float apq = A[p * n + q];
                if (fabsf(apq) < 1e-20f) continue;
                float theta = 0.5f * (A[q * n + q] - A[p * n + p]) / apq;
                float tt = 1.f / (fabsf(theta) + sqrtf(theta * theta + 1.f));
                if (theta < 0.f) tt = -tt;
                float c = 1.f / sqrtf(tt * tt + 1.f);
                float sn = tt * c;
                for (int k = 0; k < n; ++k) {      // rows p,q  (A <- R A)
                    float t1 = A[p * n + k], t2 = A[q * n + k];
                    A[p * n + k] = c * t1 - sn * t2;
                    A[q * n + k] = sn * t1 + c * t2;
                }
                for (int k = 0; k < n; ++k) {      // cols p,q  (A <- A R^T)
                    float t1 = A[k * n + p], t2 = A[k * n + q];
                    A[k * n + p] = c * t1 - sn * t2;
                    A[k * n + q] = sn * t1 + c * t2;
                }
                for (int k = 0; k < n; ++k) {      // V <- V R^T
                    float t1 = V[k * n + p], t2 = V[k * n + q];
                    V[k * n + p] = c * t1 - sn * t2;
                    V[k * n + q] = sn * t1 + c * t2;
                }
            }
}

__global__ void solve_kernel(const float* __restrict__ xtx,
                             const float* __restrict__ tw,
                             float* __restrict__ outF) {
    int b = threadIdx.x;
    if (b >= NB) return;
    float A[81], V[81];
    for (int i = 0; i < 81; ++i) A[i] = xtx[b * 81 + i];
    jacobi_sym(A, V, 9, 16);
    int idx = 0; float mn = A[0];
    for (int i = 1; i < 9; ++i) { float d = A[i * 9 + i]; if (d < mn) { mn = d; idx = i; } }
    float F[9];
    for (int i = 0; i < 9; ++i) F[i] = V[i * 9 + idx];   // smallest right singular vector
    float M3[9], V3[9];
    for (int i = 0; i < 3; ++i)
        for (int j = 0; j < 3; ++j) {
            float a = 0.f;
            for (int k = 0; k < 3; ++k) a += F[i * 3 + k] * F[j * 3 + k];
            M3[i * 3 + j] = a + ((i == j) ? 1e-4f : 0.f);
        }
    jacobi_sym(M3, V3, 3, 12);
    int i3 = 0; float mn3 = M3[0];
    for (int i = 1; i < 3; ++i) { float d = M3[i * 3 + i]; if (d < mn3) { mn3 = d; i3 = i; } }
    float u[3] = { V3[0 * 3 + i3], V3[1 * 3 + i3], V3[2 * 3 + i3] };
    float g[3];
    for (int c = 0; c < 3; ++c) g[c] = u[0] * F[c] + u[1] * F[3 + c] + u[2] * F[6 + c];
    float Fp[9];
    for (int r = 0; r < 3; ++r)
        for (int c = 0; c < 3; ++c) Fp[r * 3 + c] = F[r * 3 + c] - u[r] * g[c];
    const float* t = tw + b * 8;
    float s1 = t[0], c1x = t[1], c1y = t[2];
    float s2 = t[3], c2x = t[4], c2y = t[5];
    float t1[9] = { s1, 0.f, -c1x * s1, 0.f, s1, -c1y * s1, 0.f, 0.f, 1.f };
    float t2[9] = { s2, 0.f, -c2x * s2, 0.f, s2, -c2y * s2, 0.f, 0.f, 1.f };
    float tmp[9];
    for (int r = 0; r < 3; ++r)
        for (int c = 0; c < 3; ++c) {
            float a = 0.f;
            for (int k = 0; k < 3; ++k) a += Fp[r * 3 + k] * t2[k * 3 + c];
            tmp[r * 3 + c] = a;
        }
    for (int r = 0; r < 3; ++r)
        for (int c = 0; c < 3; ++c) {
            float a = 0.f;
            for (int k = 0; k < 3; ++k) a += t1[k * 3 + r] * tmp[k * 3 + c]; // t1^T
            outF[b * 9 + r * 3 + c] = a;
        }
}

// ---------------------------------------------------------------------------
// workspace layout (bytes) — ~195 MB total, ping/pong bf16 activations
// ---------------------------------------------------------------------------
constexpr size_t OFF_WB1  = 0;                                   // 128*64 bf16
constexpr size_t OFF_WB2  = OFF_WB1 + (size_t)128 * 64 * 2;      // 1024*128
constexpr size_t OFF_WB3  = OFF_WB2 + (size_t)1024 * 128 * 2;    // 512*1024
constexpr size_t OFF_WB4  = OFF_WB3 + (size_t)512 * 1024 * 2;    // 256*512
constexpr size_t OFF_P1   = OFF_WB4 + (size_t)256 * 512 * 2;     // pts1n f32
constexpr size_t OFF_P2   = OFF_P1 + (size_t)MT * 2 * 4;         // pts2n f32
constexpr size_t OFF_PING = OFF_P2 + (size_t)MT * 2 * 4;         // <=65536x1024 bf16
constexpr size_t OFF_PONG = OFF_PING + (size_t)MT * 1024 * 2;    // <=65536x512 bf16
constexpr size_t OFF_LOG  = OFF_PONG + (size_t)MT * 512 * 2;     // logits f32
constexpr size_t OFF_TW   = OFF_LOG + (size_t)MT * 4;            // 16x8 f32
constexpr size_t OFF_XTX  = OFF_TW + (size_t)NB * 8 * 4;         // 16x81 f32

extern "C" void kernel_launch(void* const* d_in, const int* in_sizes, int n_in,
                              void* d_out, int out_size, void* d_ws, size_t ws_size,
                              hipStream_t stream) {
    const float* pts  = (const float*)d_in[0];
    const float* side = (const float*)d_in[1];
    const float* w0 = (const float*)d_in[2];  const float* b0 = (const float*)d_in[3];
    const float* w1 = (const float*)d_in[4];
    const float* w2 = (const float*)d_in[6];
    const float* w3 = (const float*)d_in[8];
    const float* w4 = (const float*)d_in[10];
    const float* w5 = (const float*)d_in[12]; const float* b5 = (const float*)d_in[13];
    const float* g0 = (const float*)d_in[14]; const float* be0 = (const float*)d_in[15];
    const float* g1 = (const float*)d_in[16]; const float* be1 = (const float*)d_in[17];
    const float* g2 = (const float*)d_in[18]; const float* be2 = (const float*)d_in[19];
    const float* g3 = (const float*)d_in[20]; const float* be3 = (const float*)d_in[21];
    const float* g4 = (const float*)d_in[22]; const float* be4 = (const float*)d_in[23];

    char* ws = (char*)d_ws;
    __bf16* wb1 = (__bf16*)(ws + OFF_WB1);
    __bf16* wb2 = (__bf16*)(ws + OFF_WB2);
    __bf16* wb3 = (__bf16*)(ws + OFF_WB3);
    __bf16* wb4 = (__bf16*)(ws + OFF_WB4);
    float*  p1n = (float*)(ws + OFF_P1);
    float*  p2n = (float*)(ws + OFF_P2);
    __bf16* ping = (__bf16*)(ws + OFF_PING);
    __bf16* pong = (__bf16*)(ws + OFF_PONG);
    float*  logits = (float*)(ws + OFF_LOG);
    float*  tw     = (float*)(ws + OFF_TW);
    float*  xtx    = (float*)(ws + OFF_XTX);

    float* out = (float*)d_out;           // out(144) | T1(144) | T2(144) | weights(65536)
    float* outT1 = out + 144;
    float* outT2 = out + 288;
    float* outW  = out + 432;

    // weight conversion
    cvt_bf16_kernel<<<(128 * 64 + 255) / 256, 256, 0, stream>>>(w1, wb1, 128 * 64);
    cvt_bf16_kernel<<<(1024 * 128 + 255) / 256, 256, 0, stream>>>(w2, wb2, 1024 * 128);
    cvt_bf16_kernel<<<(512 * 1024 + 255) / 256, 256, 0, stream>>>(w3, wb3, 512 * 1024);
    cvt_bf16_kernel<<<(256 * 512 + 255) / 256, 256, 0, stream>>>(w4, wb4, 256 * 512);

    // plain normalization, T1/T2
    prep_kernel<<<NB, 256, 0, stream>>>(pts, p1n, p2n, outT1, outT2);

    // MLP: activations ping/pong in bf16; wave = 32x64 output region
    layer0_kernel<<<MT / 256, 256, 0, stream>>>(p1n, p2n, side, w0, b0, ping);
    instnorm_kernel<<<NB * (64 / 16), 256, 0, stream>>>(ping, g0, be0, 64);

    gemm_bf16_kernel<<<(MT / 32) * (128 / 64) / 8, 256, 0, stream>>>(ping, wb1, pong, 64, 128);
    instnorm_kernel<<<NB * (128 / 16), 256, 0, stream>>>(pong, g1, be1, 128);

    gemm_bf16_kernel<<<(MT / 32) * (1024 / 64) / 8, 256, 0, stream>>>(pong, wb2, ping, 128, 1024);
    instnorm_kernel<<<NB * (1024 / 16), 256, 0, stream>>>(ping, g2, be2, 1024);

    gemm_bf16_kernel<<<(MT / 32) * (512 / 64) / 8, 256, 0, stream>>>(ping, wb3, pong, 1024, 512);
    instnorm_kernel<<<NB * (512 / 16), 256, 0, stream>>>(pong, g3, be3, 512);

    gemm_bf16_kernel<<<(MT / 32) * (256 / 64) / 8, 256, 0, stream>>>(pong, wb4, ping, 512, 256);
    instnorm_kernel<<<NB * (256 / 16), 256, 0, stream>>>(ping, g4, be4, 256);

    layer5_kernel<<<MT / 8, 256, 0, stream>>>(ping, w5, b5, logits);
    softmax_kernel<<<NB, 256, 0, stream>>>(logits, outW);

    // weighted normalization + 9x9 / 3x3 solves
    wnorm_kernel<<<NB, 256, 0, stream>>>(p1n, p2n, outW, tw);
    xtx_kernel<<<NB, 256, 0, stream>>>(p1n, p2n, outW, tw, xtx);
    solve_kernel<<<1, 32, 0, stream>>>(xtx, tw, out);
}
